// Tiny_ConvCIFAR10_KAN_Discriminator_33921651704119
// MI455X (gfx1250) — matrix operand order repaired and checked by reading, not compiled, not run on hardware
//
#include <hip/hip_runtime.h>
#include <hip/hip_fp16.h>
#include <math.h>

typedef __attribute__((ext_vector_type(16))) _Float16 v16h;
typedef __attribute__((ext_vector_type(8)))  float    v8f;

#define KFEAT 288   // 48 relu features + 48*5 spline-basis features
#define NPAD  16    // out channels padded 8 -> 16 for WMMA
#define NK_CH 9     // 288 / 32 K-chunks
#define IMGSZ (3 * 32 * 32)

// ---------------------------------------------------------------------------
// Kernel W: build transposed combined weight WcT[16][288] in f16.
//   column n<8  -> out channel o=n ; n>=8 -> zero pad
//   k < 48      -> base_weight[o][k]
//   k >= 48     -> spline_weight[o][(k-48)/5][(k-48)%5]
// ---------------------------------------------------------------------------
__global__ void kan_build_weightT(const float* __restrict__ base_w,    // [8][48]
                                  const float* __restrict__ spline_w,  // [8][48][5]
                                  _Float16* __restrict__ WcT)          // [16][288]
{
    for (int i = threadIdx.x; i < NPAD * KFEAT; i += blockDim.x) {
        int n = i / KFEAT, k = i - n * KFEAT;
        float v = 0.f;
        if (n < 8) {
            if (k < 48) {
                v = base_w[n * 48 + k];
            } else {
                int idx = k - 48;
                int f = idx / 5;
                int j = idx - 5 * f;
                v = spline_w[(n * 48 + f) * 5 + j];
            }
        }
        WcT[n * KFEAT + k] = (_Float16)v;
    }
}

// ---------------------------------------------------------------------------
// Uniform quadratic B-spline, knots at integers, support [0,3).
// Matches Cox-de Boor (order 2) on the uniform extended grid.
// ---------------------------------------------------------------------------
__device__ __forceinline__ float quad_bspline(float u) {
    if (u <= 0.f || u >= 3.f) return 0.f;
    if (u < 1.f)  return 0.5f * u * u;
    if (u < 2.f)  return 0.5f * (-2.f * u * u + 6.f * u - 3.f);
    float v = 3.f - u;
    return 0.5f * v * v;
}

// ---------------------------------------------------------------------------
// Kernel A: per-image pooled augmented features.
// Block = one image, 256 threads = 256 output pixels (16x16, stride 2, pad 1).
// Image (12 KB) is staged once into LDS with coalesced float4 loads, killing
// the ~4x global-read redundancy of overlapping 4x4/stride-2 patches.
// acc[k] = sum over pixels of augmented feature k ; write mean as f16.
// ---------------------------------------------------------------------------
__global__ void kan_pool_features(const float* __restrict__ x,  // [B][3][32][32]
                                  _Float16* __restrict__ Pf)    // [B][288]
{
    __shared__ float img[IMGSZ];     // 12288 B
    __shared__ float acc[KFEAT];     //  1152 B
    const int b    = blockIdx.x;
    const int tid  = threadIdx.x;    // pixel id 0..255
    const int lane = tid & 31;

    // stage image: 3072 floats = 768 float4 ; 3 vec4 loads per thread
    {
        const float4* src = (const float4*)(x + (size_t)b * IMGSZ);
        float4* dst = (float4*)img;
        #pragma unroll
        for (int i = 0; i < 3; ++i)
            dst[tid + 256 * i] = src[tid + 256 * i];
    }
    for (int k = tid; k < KFEAT; k += 256) acc[k] = 0.f;
    __syncthreads();

    const int ho = tid >> 4, wo = tid & 15;

    for (int f = 0; f < 48; ++f) {           // feature order (C, kh, kw)
        int c  = f >> 4;
        int kh = (f >> 2) & 3;
        int kw = f & 3;
        int hi = ho * 2 - 1 + kh;
        int wi = wo * 2 - 1 + kw;
        float p = 0.f;
        if ((unsigned)hi < 32u && (unsigned)wi < 32u)
            p = img[(c * 32 + hi) * 32 + wi];

        float v[6];
        v[0] = p > 0.f ? p : 0.f;            // ReLU base branch
        // knots t_j = (j-2)*h - 1, h = 2/3 ; u_j = (p - t_j)/h = 1.5*(p+1) + 2 - j
        float t = (p + 1.f) * 1.5f;
        #pragma unroll
        for (int j = 0; j < 5; ++j) v[1 + j] = quad_bspline(t + 2.f - (float)j);

        // wave32 butterfly reduction of the 6 values
        #pragma unroll
        for (int j = 0; j < 6; ++j) {
            float s = v[j];
            for (int m = 16; m >= 1; m >>= 1) s += __shfl_xor(s, m, 32);
            v[j] = s;
        }
        if (lane == 0) {
            atomicAdd(&acc[f], v[0]);
            #pragma unroll
            for (int j = 0; j < 5; ++j)
                atomicAdd(&acc[48 + f * 5 + j], v[1 + j]);
        }
    }
    __syncthreads();

    const float inv = 1.f / 256.f;           // AdaptiveAvgPool mean
    for (int k = tid; k < KFEAT; k += 256)
        Pf[(size_t)b * KFEAT + k] = (_Float16)(acc[k] * inv);
}

// ---------------------------------------------------------------------------
// Kernel B: [B x 288] (f16) x WcT [16 x 288] (f16) via v_wmma_f32_16x16x32_f16,
// then lin_w dot + sigmoid. One wave per 16 images. 9 WMMAs per wave.
// ---------------------------------------------------------------------------
__global__ void kan_gemm_head(const _Float16* __restrict__ Pf,   // [B][288]
                              const _Float16* __restrict__ WcT,  // [16][288]
                              const float* __restrict__ lin_w,   // [8]
                              const float* __restrict__ lin_b,   // [1]
                              float* __restrict__ out)           // [B]
{
    const int lane = threadIdx.x & 31;
    const int wave = blockIdx.x * (blockDim.x >> 5) + (threadIdx.x >> 5);
    const int m    = lane & 15;   // row (A) / column (B) index
    const int hs   = lane >> 4;   // half-wave select

    // B-matrix regs: lane holds column n=m; K slice [kc*32 + 16*hs, +16)
    v16h breg[NK_CH];
    const _Float16* bcol = WcT + m * KFEAT;
    #pragma unroll
    for (int kc = 0; kc < NK_CH; ++kc)
        breg[kc] = *(const v16h*)(bcol + kc * 32 + 16 * hs);

    const int row = wave * 16 + m;
    const _Float16* arow = Pf + (size_t)row * KFEAT;

    v8f c = {};
    #pragma unroll
    for (int kc = 0; kc < NK_CH; ++kc) {
        // A uses contiguous k order here: lane M=m holds K in [16*hs,16*hs+16) (matches B)
        union { v16h v; float4 f4[2]; } a;
        a.f4[0] = *(const float4*)(arow + kc * 32 + 16 * hs);
        a.f4[1] = *(const float4*)(arow + kc * 32 + 16 * hs + 8);
        c = __builtin_amdgcn_wmma_f32_16x16x32_f16(false, a.v, false, breg[kc],
                                                   (short)0, c, false, false);
    }

    // Epilogue: out[row m'] = sigmoid( sum_o D[m'][o]*lin_w[o] + lin_b )
    // D layout: VGPR r holds M=r (lanes 0-15) / M=8+r (lanes 16-31), N=lane&15.
    float w  = (m < 8) ? lin_w[m] : 0.f;
    float bb = lin_b[0];
    #pragma unroll
    for (int r = 0; r < 8; ++r) {
        float s = c[r] * w;
        #pragma unroll
        for (int mm = 1; mm < 16; mm <<= 1) s += __shfl_xor(s, mm, 32);
        if (lane == 0)
            out[wave * 16 + r]     = 1.f / (1.f + __expf(-(s + bb)));
        if (lane == 16)
            out[wave * 16 + 8 + r] = 1.f / (1.f + __expf(-(s + bb)));
    }
}

// ---------------------------------------------------------------------------
extern "C" void kernel_launch(void* const* d_in, const int* in_sizes, int n_in,
                              void* d_out, int out_size, void* d_ws, size_t ws_size,
                              hipStream_t stream) {
    const float* x   = (const float*)d_in[0];  // [B,3,32,32]
    const float* bw  = (const float*)d_in[1];  // [8,48]
    const float* sw  = (const float*)d_in[2];  // [8,48,5]
    const float* lw  = (const float*)d_in[3];  // [1,8]
    const float* lb  = (const float*)d_in[4];  // [1]
    float* out = (float*)d_out;

    const int B = in_sizes[0] / IMGSZ;         // 2048

    // workspace layout: Pf [B][288] f16, then WcT [16][288] f16 (32B aligned)
    _Float16* Pf  = (_Float16*)d_ws;
    _Float16* WcT = (_Float16*)((char*)d_ws + (size_t)B * KFEAT * sizeof(_Float16));

    kan_build_weightT<<<1, 256, 0, stream>>>(bw, sw, WcT);
    kan_pool_features<<<B, 256, 0, stream>>>(x, Pf);

    const int waves  = B / 16;          // 128
    const int blocks = waves / 8;       // 8 waves per 256-thread block
    kan_gemm_head<<<blocks, 256, 0, stream>>>(Pf, WcT, lw, lb, out);
}
